// VoxelResBackBone8x_63677185130616
// MI455X (gfx1250) — matrix operand order repaired and checked by reading, not compiled, not run on hardware
//
#include <hip/hip_runtime.h>

typedef __attribute__((ext_vector_type(16))) __bf16 v16bf;
typedef __attribute__((ext_vector_type(8)))  float  v8f;

// ---------------- bf16 helpers (ushort storage, __bf16 only in device code) ----
__device__ __forceinline__ unsigned short f2bf(float f) {
    unsigned int u = __float_as_uint(f);
    unsigned int r = (u + 0x7fffu + ((u >> 16) & 1u)) >> 16;   // RNE
    return (unsigned short)r;
}
__device__ __forceinline__ float bf2f(unsigned short s) {
    return __uint_as_float(((unsigned int)s) << 16);
}

union BFrag {                       // 32 bytes: one WMMA bf16 operand per lane
    v16bf v;
    unsigned short u[16];
    uint2 d[4];
    uint4 q[2];
};
union Pack8 {                       // 16 bytes: 8 consecutive bf16 channels
    uint4 q;
    unsigned short u[8];
};

// ---------------- scatter voxels into dense bf16 grid + mask --------------------
__global__ void k_scatter(const float* __restrict__ vf, const int* __restrict__ vc,
                          unsigned short* __restrict__ dense, unsigned char* __restrict__ mask,
                          int N, int Dd, int Hh, int Ww) {
    int i = blockIdx.x * blockDim.x + threadIdx.x;
    if (i >= N) return;
    int b = vc[4*i+0], z = vc[4*i+1], y = vc[4*i+2], x = vc[4*i+3];
    long pos = (((long)b * Dd + z) * Hh + y) * Ww + x;
    mask[pos] = 1;
    union { uint2 d; unsigned short u[4]; } o;
    for (int c = 0; c < 4; ++c) o.u[c] = f2bf(vf[4*i + c]);
    *(uint2*)(dense + pos*4) = o.d;
}

// ------ weight repack: OIDHW f32 -> [Co][Kpad] bf16, zero-padded past Ktot ------
__global__ void k_repack_w(const float* __restrict__ w, unsigned short* __restrict__ out,
                           int Co, int Ci, int kd, int kh, int kw, int Kpad) {
    int tot = Co * Kpad;
    int i = blockIdx.x * blockDim.x + threadIdx.x;
    if (i >= tot) return;
    int Ktot = kd*kh*kw*Ci;
    int o = i / Kpad, k = i - o*Kpad;
    unsigned short val = 0;
    if (k < Ktot) {
        int tap = k / Ci, c = k - tap*Ci;
        int khw = kh*kw;
        int dz = tap / khw; int t2 = tap - dz*khw;
        int dy = t2 / kw;   int dx = t2 - dy*kw;
        val = f2bf(w[(((long)(o*Ci + c) * kd + dz) * kh + dy) * kw + dx]);
    }
    out[i] = val;
}

// ---------------- fold BN (+ optional conv bias) into scale/shift ---------------
__global__ void k_bn_prep(const float* __restrict__ g, const float* __restrict__ b,
                          const float* __restrict__ m, const float* __restrict__ v,
                          const float* __restrict__ bias,
                          float* __restrict__ scale, float* __restrict__ shift, int C) {
    int i = blockIdx.x * blockDim.x + threadIdx.x;
    if (i >= C) return;
    float s = g[i] * rsqrtf(v[i] + 1e-3f);
    scale[i] = s;
    shift[i] = b[i] - m[i]*s + (bias ? bias[i]*s : 0.f);
}

// ---------------- mask dilation for strided convs ------------------------------
__global__ void k_mask_down(const unsigned char* __restrict__ mi, unsigned char* __restrict__ mo,
                            int Di,int Hi,int Wi, int Do,int Ho,int Wo,
                            int kd,int kh,int kw, int sd,int sh,int sw,
                            int pd,int ph,int pw, int nvox) {
    int vo = blockIdx.x * blockDim.x + threadIdx.x;
    if (vo >= nvox) return;
    int hw = Ho*Wo, dhw = Do*hw;
    int b = vo / dhw; int r = vo - b*dhw;
    int zo = r / hw;  r -= zo*hw;
    int yo = r / Wo;  int xo = r - yo*Wo;
    unsigned char mv = 0;
    for (int dz = 0; dz < kd; ++dz) { int zi = zo*sd - pd + dz; if (zi < 0 || zi >= Di) continue;
      for (int dy = 0; dy < kh; ++dy) { int yi = yo*sh - ph + dy; if (yi < 0 || yi >= Hi) continue;
        for (int dx = 0; dx < kw; ++dx) { int xi = xo*sw - pw + dx; if (xi < 0 || xi >= Wi) continue;
            if (mi[(((long)b*Di + zi)*Hi + yi)*Wi + xi]) mv = 1;
        }}}
    mo[vo] = mv;
}

// ---------------- generic masked conv3d as implicit GEMM via WMMA bf16 ---------
// One wave: 16 out-channels x 16 voxels; K = taps*Ci in chunks of 32 (Kpad-padded).
// Ci is a power of two (ciShift); chunk segments are contiguous channel runs ->
// b128/b64 vector loads. Tap (dz,dy,dx) decode is a small LDS table.
__global__ void k_conv_wmma(
    const unsigned short* __restrict__ xin,      // (B,Di,Hi,Wi,Ci) bf16
    const unsigned char*  __restrict__ omask,    // output-resolution mask
    const unsigned short* __restrict__ wbuf,     // [Co][Kpad] bf16 (zero padded)
    const float* __restrict__ scale, const float* __restrict__ shift,
    const unsigned short* resid,                 // nullable (B,Do,Ho,Wo,Co) bf16
    unsigned short* ybf,                         // (B,Do,Ho,Wo,Co) bf16, or null
    float* yf32,                                 // NCDHW f32, or null
    int Di,int Hi,int Wi,int Ci, int Do,int Ho,int Wo,int Co,
    int kd,int kh,int kw, int sd,int sh,int sw, int pd,int ph,int pw,
    int ciShift, int Kpad, int nvox)
{
    __shared__ int s_tap[32];
    const int ntaps = kd*kh*kw;
    if (threadIdx.x < 32) {
        int tap = threadIdx.x;
        int khw = kh*kw;
        int dz = tap / khw; int t2 = tap - dz*khw;
        int dy = t2 / kw;   int dx = t2 - dy*kw;
        s_tap[tap] = (dz << 20) | (dy << 10) | dx;
    }
    __syncthreads();

    const int lane = threadIdx.x & 31;
    const int wave = threadIdx.x >> 5;
    const int half = lane >> 4;
    const int l15  = lane & 15;
    const int vt = (blockIdx.x * (blockDim.x >> 5) + wave) * 16;
    if (vt >= nvox) return;
    const int ct = blockIdx.y * 16;

    int vo = vt + l15;
    bool vok = vo < nvox;
    int vv = vok ? vo : 0;
    int hw = Ho*Wo, dhw = Do*hw;
    int b  = vv / dhw; int r = vv - b*dhw;
    int zo = r / hw;   r -= zo*hw;
    int yo = r / Wo;   int xo = r - yo*Wo;
    int z0 = zo*sd - pd, y0 = yo*sh - ph, x0 = xo*sw - pw;

    const int nk = Kpad >> 5;
    const unsigned ciMask = (unsigned)Ci - 1u;

    v8f acc = {};
    // tile-level sparsity skip (dominant win at ~1.8% occupancy in stage 1)
    bool active = vok && (omask[vv] != 0);
    unsigned long long ball = __ballot(active);

    if (ball != 0ULL) {
        const unsigned short* wrow = wbuf + (long)(ct + l15) * Kpad;
        for (int kk = 0; kk < nk; ++kk) {
            const int kb = kk << 5;
            const int kbase = kb + half*16;
            BFrag bf;
            if (Ci >= 16) {
                // one tap, 16 consecutive channels -> 2x b128
                int tap = kbase >> ciShift;
                int c0  = kbase & ciMask;
                uint4 q0 = make_uint4(0,0,0,0), q1 = q0;
                if (vok && tap < ntaps) {
                    int t  = s_tap[tap];
                    int zi = z0 + (t >> 20), yi = y0 + ((t >> 10) & 1023), xi = x0 + (t & 1023);
                    if ((unsigned)zi < (unsigned)Di && (unsigned)yi < (unsigned)Hi &&
                        (unsigned)xi < (unsigned)Wi) {
                        const uint4* p = (const uint4*)(xin +
                            ((((long)b*Di + zi)*Hi + yi)*Wi + xi) * (long)Ci + c0);
                        q0 = p[0]; q1 = p[1];
                    }
                }
                bf.q[0] = q0; bf.q[1] = q1;
            } else {
                // Ci == 4: 4 taps x 4 channels -> 4x b64
                #pragma unroll
                for (int s = 0; s < 4; ++s) {
                    int k0 = kbase + (s << 2);
                    int tap = k0 >> 2;
                    uint2 d = make_uint2(0,0);
                    if (vok && tap < ntaps) {
                        int t  = s_tap[tap];
                        int zi = z0 + (t >> 20), yi = y0 + ((t >> 10) & 1023), xi = x0 + (t & 1023);
                        if ((unsigned)zi < (unsigned)Di && (unsigned)yi < (unsigned)Hi &&
                            (unsigned)xi < (unsigned)Wi)
                            d = *(const uint2*)(xin + ((((long)b*Di + zi)*Hi + yi)*Wi + xi) * 4);
                    }
                    bf.d[s] = d;
                }
            }
            // A fragment: 16 contiguous k at kb + half*16 (zero-padded wbuf -> no guards)
            BFrag af;
            af.q[0] = *(const uint4*)(wrow + kb + half*16);
            af.q[1] = *(const uint4*)(wrow + kb + half*16 + 8);
            acc = __builtin_amdgcn_wmma_f32_16x16x32_bf16(
                false, af.v, false, bf.v, (short)0, acc, false, false);
        }
    }

    if (!vok) return;
    float mv = active ? 1.0f : 0.0f;
    long obase = (long)vo * Co + ct + half*8;      // 8 consecutive channels
    float rv[8];
    if (resid) {
        Pack8 rp; rp.q = *(const uint4*)(resid + obase);
        #pragma unroll
        for (int rr = 0; rr < 8; ++rr) rv[rr] = bf2f(rp.u[rr]);
    } else {
        #pragma unroll
        for (int rr = 0; rr < 8; ++rr) rv[rr] = 0.f;
    }
    if (ybf) {
        Pack8 op;
        #pragma unroll
        for (int rr = 0; rr < 8; ++rr) {
            int co = ct + half*8 + rr;             // D layout: M = half*8 + rr, N = l15
            float out = acc[rr] * scale[co] + shift[co] + rv[rr];
            op.u[rr] = f2bf(fmaxf(out, 0.0f) * mv);
        }
        *(uint4*)(ybf + obase) = op.q;
    } else {
        #pragma unroll
        for (int rr = 0; rr < 8; ++rr) {
            int co = ct + half*8 + rr;
            float out = acc[rr] * scale[co] + shift[co] + rv[rr];
            out = fmaxf(out, 0.0f) * mv;
            long idx = (((long)b * Co + co) * Do + zo) * hw + (long)yo * Wo + xo;
            yf32[idx] = out;
        }
    }
}

// ------------------------------- host side -------------------------------------
static inline int cdiv(int a, int b) { return (a + b - 1) / b; }
static inline int ilog2(int x) { int s = 0; while ((1 << s) < x) ++s; return s; }

extern "C" void kernel_launch(void* const* d_in, const int* in_sizes, int n_in,
                              void* d_out, int out_size, void* d_ws, size_t ws_size,
                              hipStream_t stream) {
    (void)in_sizes; (void)n_in; (void)out_size; (void)ws_size;
    const int BB = 2, D0 = 41, H0 = 200, W0 = 200, NV = 60000;
    const int nv1 = BB*41*200*200, nv2 = BB*21*100*100, nv3 = BB*11*50*50,
              nv4 = BB*5*25*25,   nv5 = BB*2*25*25;

    // ---- input walk (setup_inputs insertion order) ----
    struct Conv { const float *w, *bias, *g, *b, *m, *v; };
    int idx = 0;
    const float* vf = (const float*)d_in[idx++];
    const int*   vc = (const int*)d_in[idx++];
    idx++; // batch_size (known == 2)
    Conv cv[21]; int ci = 0;
    auto take_bn = [&](Conv& c) {
        c.g = (const float*)d_in[idx++]; c.b = (const float*)d_in[idx++];
        c.m = (const float*)d_in[idx++]; c.v = (const float*)d_in[idx++];
    };
    auto take_plain = [&]() { cv[ci].w = (const float*)d_in[idx++]; cv[ci].bias = nullptr; take_bn(cv[ci]); ci++; };
    auto take_block = [&]() {
        for (int j = 0; j < 2; ++j) {
            cv[ci].w = (const float*)d_in[idx++];
            cv[ci].bias = (const float*)d_in[idx++];
            take_bn(cv[ci]); ci++;
        }
    };
    take_plain();                 // conv_input          -> cv[0]
    take_block(); take_block();   // conv1 blocks        -> cv[1..4]
    take_plain();                 // conv2 down          -> cv[5]
    take_block(); take_block();   //                     -> cv[6..9]
    take_plain();                 // conv3 down          -> cv[10]
    take_block(); take_block();   //                     -> cv[11..14]
    take_plain();                 // conv4 down          -> cv[15]
    take_block(); take_block();   //                     -> cv[16..19]
    take_plain();                 // conv_out            -> cv[20]

    // ---- workspace layout ----
    char* ws = (char*)d_ws; size_t off = 0;
    auto alloc = [&](size_t bytes) -> void* {
        void* p = ws + off; off = (off + bytes + 255) & ~(size_t)255; return p;
    };
    unsigned short* in_dense = (unsigned short*)alloc((size_t)nv1*4*2);
    unsigned char*  mask1    = (unsigned char*) alloc(nv1);
    unsigned short* a1A = (unsigned short*)alloc((size_t)nv1*16*2);
    unsigned short* a1B = (unsigned short*)alloc((size_t)nv1*16*2);
    unsigned char*  mask2 = (unsigned char*)alloc(nv2);
    unsigned short* a2A = (unsigned short*)alloc((size_t)nv2*32*2);
    unsigned short* a2B = (unsigned short*)alloc((size_t)nv2*32*2);
    unsigned char*  mask3 = (unsigned char*)alloc(nv3);
    unsigned short* a3A = (unsigned short*)alloc((size_t)nv3*64*2);
    unsigned short* a3B = (unsigned short*)alloc((size_t)nv3*64*2);
    unsigned char*  mask4 = (unsigned char*)alloc(nv4);
    unsigned short* a4A = (unsigned short*)alloc((size_t)nv4*128*2);
    unsigned short* a4B = (unsigned short*)alloc((size_t)nv4*128*2);
    unsigned char*  mask5 = (unsigned char*)alloc(nv5);
    unsigned short* wbuf  = (unsigned short*)alloc((size_t)128*3456*2);
    float* scl = (float*)alloc(128*4);
    float* sht = (float*)alloc(128*4);

    hipMemsetAsync(in_dense, 0, (size_t)nv1*4*2, stream);
    hipMemsetAsync(mask1, 0, (size_t)nv1, stream);
    k_scatter<<<cdiv(NV,256), 256, 0, stream>>>(vf, vc, in_dense, mask1, NV, D0, H0, W0);

    auto run_conv = [&](const Conv& c, const unsigned short* xin, const unsigned char* om,
                        const unsigned short* resid, unsigned short* ybf, float* yf32,
                        int Di,int Hi,int Wi,int Ci, int Do,int Ho,int Wo,int Co,
                        int kd,int kh,int kw, int sd,int sh,int sw, int pd,int ph,int pw) {
        int Ktot = kd*kh*kw*Ci;
        int Kpad = cdiv(Ktot, 32) * 32;
        int ciShift = ilog2(Ci);
        k_repack_w<<<cdiv(Co*Kpad,256), 256, 0, stream>>>(c.w, wbuf, Co, Ci, kd, kh, kw, Kpad);
        k_bn_prep<<<1, 256, 0, stream>>>(c.g, c.b, c.m, c.v, c.bias, scl, sht, Co);
        int nv = BB*Do*Ho*Wo;
        dim3 grid(cdiv(nv, 16*4), Co/16);
        k_conv_wmma<<<grid, 128, 0, stream>>>(xin, om, wbuf, scl, sht, resid, ybf, yf32,
            Di,Hi,Wi,Ci, Do,Ho,Wo,Co, kd,kh,kw, sd,sh,sw, pd,ph,pw, ciShift, Kpad, nv);
    };
    auto run_mask = [&](const unsigned char* mi, unsigned char* mo,
                        int Di,int Hi,int Wi,int Do,int Ho,int Wo,
                        int kd,int kh,int kw,int sd,int sh,int sw,int pd,int ph,int pw) {
        int nv = BB*Do*Ho*Wo;
        k_mask_down<<<cdiv(nv,256), 256, 0, stream>>>(mi, mo, Di,Hi,Wi, Do,Ho,Wo,
            kd,kh,kw, sd,sh,sw, pd,ph,pw, nv);
    };

    // stage 1: 41x200x200, C 4->16, subm convs
    run_conv(cv[0], in_dense, mask1, nullptr, a1A, nullptr, 41,200,200,4,  41,200,200,16, 3,3,3, 1,1,1, 1,1,1);
    run_conv(cv[1], a1A, mask1, nullptr, a1B, nullptr, 41,200,200,16, 41,200,200,16, 3,3,3, 1,1,1, 1,1,1);
    run_conv(cv[2], a1B, mask1, a1A,     a1A, nullptr, 41,200,200,16, 41,200,200,16, 3,3,3, 1,1,1, 1,1,1);
    run_conv(cv[3], a1A, mask1, nullptr, a1B, nullptr, 41,200,200,16, 41,200,200,16, 3,3,3, 1,1,1, 1,1,1);
    run_conv(cv[4], a1B, mask1, a1A,     a1A, nullptr, 41,200,200,16, 41,200,200,16, 3,3,3, 1,1,1, 1,1,1);

    // stage 2: down to 21x100x100, C 16->32
    run_mask(mask1, mask2, 41,200,200, 21,100,100, 3,3,3, 2,2,2, 1,1,1);
    run_conv(cv[5], a1A, mask2, nullptr, a2A, nullptr, 41,200,200,16, 21,100,100,32, 3,3,3, 2,2,2, 1,1,1);
    run_conv(cv[6], a2A, mask2, nullptr, a2B, nullptr, 21,100,100,32, 21,100,100,32, 3,3,3, 1,1,1, 1,1,1);
    run_conv(cv[7], a2B, mask2, a2A,     a2A, nullptr, 21,100,100,32, 21,100,100,32, 3,3,3, 1,1,1, 1,1,1);
    run_conv(cv[8], a2A, mask2, nullptr, a2B, nullptr, 21,100,100,32, 21,100,100,32, 3,3,3, 1,1,1, 1,1,1);
    run_conv(cv[9], a2B, mask2, a2A,     a2A, nullptr, 21,100,100,32, 21,100,100,32, 3,3,3, 1,1,1, 1,1,1);

    // stage 3: down to 11x50x50, C 32->64
    run_mask(mask2, mask3, 21,100,100, 11,50,50, 3,3,3, 2,2,2, 1,1,1);
    run_conv(cv[10], a2A, mask3, nullptr, a3A, nullptr, 21,100,100,32, 11,50,50,64, 3,3,3, 2,2,2, 1,1,1);
    run_conv(cv[11], a3A, mask3, nullptr, a3B, nullptr, 11,50,50,64, 11,50,50,64, 3,3,3, 1,1,1, 1,1,1);
    run_conv(cv[12], a3B, mask3, a3A,     a3A, nullptr, 11,50,50,64, 11,50,50,64, 3,3,3, 1,1,1, 1,1,1);
    run_conv(cv[13], a3A, mask3, nullptr, a3B, nullptr, 11,50,50,64, 11,50,50,64, 3,3,3, 1,1,1, 1,1,1);
    run_conv(cv[14], a3B, mask3, a3A,     a3A, nullptr, 11,50,50,64, 11,50,50,64, 3,3,3, 1,1,1, 1,1,1);

    // stage 4: down to 5x25x25 (pad 0,1,1), C 64->128
    run_mask(mask3, mask4, 11,50,50, 5,25,25, 3,3,3, 2,2,2, 0,1,1);
    run_conv(cv[15], a3A, mask4, nullptr, a4A, nullptr, 11,50,50,64, 5,25,25,128, 3,3,3, 2,2,2, 0,1,1);
    run_conv(cv[16], a4A, mask4, nullptr, a4B, nullptr, 5,25,25,128, 5,25,25,128, 3,3,3, 1,1,1, 1,1,1);
    run_conv(cv[17], a4B, mask4, a4A,     a4A, nullptr, 5,25,25,128, 5,25,25,128, 3,3,3, 1,1,1, 1,1,1);
    run_conv(cv[18], a4A, mask4, nullptr, a4B, nullptr, 5,25,25,128, 5,25,25,128, 3,3,3, 1,1,1, 1,1,1);
    run_conv(cv[19], a4B, mask4, a4A,     a4A, nullptr, 5,25,25,128, 5,25,25,128, 3,3,3, 1,1,1, 1,1,1);

    // conv_out: kernel (3,1,1) stride (2,1,1) pad 0 -> 2x25x25, f32 NCDHW to d_out
    run_mask(mask4, mask5, 5,25,25, 2,25,25, 3,1,1, 2,1,1, 0,0,0);
    run_conv(cv[20], a4A, mask5, nullptr, nullptr, (float*)d_out,
             5,25,25,128, 2,25,25,128, 3,1,1, 2,1,1, 0,0,0);
}
